// MultiHeadAttention_24833500905492
// MI455X (gfx1250) — compile-verified, matrix operand order checked by reading, not run on hardware
//
#include <hip/hip_runtime.h>
#include <hip/hip_bf16.h>
#include <stdint.h>

#define TT  2048
#define DDm 1024
#define HH  16
#define DHH 64

#if defined(__has_builtin)
#if __has_builtin(__builtin_amdgcn_global_load_async_to_lds_b128) && \
    __has_builtin(__builtin_amdgcn_s_wait_asynccnt)
#define HAVE_ASYNC 1
#endif
#endif
#ifndef HAVE_ASYNC
#define HAVE_ASYNC 0
#endif

typedef __attribute__((ext_vector_type(16))) _Float16 v16h;
typedef __attribute__((ext_vector_type(8)))  _Float16 v8h;
typedef __attribute__((ext_vector_type(8)))  float    v8f;
typedef __attribute__((ext_vector_type(4)))  int      v4i;

#define AS1 __attribute__((address_space(1)))
#define AS3 __attribute__((address_space(3)))

static __device__ __forceinline__ int uni(int x) {   // force wave-uniform -> SGPR
  return __builtin_amdgcn_readfirstlane(x);
}

// ---------------------------------------------------------------------------
// Fragment loaders per CDNA5 ISA 7.12.2 (wave32), 16x16x32 f16.
// ---------------------------------------------------------------------------
static __device__ __forceinline__ v16h load_frag_f16(const _Float16* __restrict__ base,
                                                     int ld, int row0, int kk, int lane) {
  const int r  = lane & 15;
  const int c0 = (lane >> 4) << 3;
  const _Float16* p = base + (size_t)(row0 + r) * ld + kk + c0;
  v8h lo = *(const v8h*)(p);
  v8h hi = *(const v8h*)(p + 16);
  v16h out;
#pragma unroll
  for (int i = 0; i < 8; ++i) { out[i] = lo[i]; out[i + 8] = hi[i]; }
  return out;
}

static __device__ __forceinline__ v16h load_frag_f32cvt(const float* __restrict__ base,
                                                        int ld, int row0, int kk, int lane) {
  const int r  = lane & 15;
  const int c0 = (lane >> 4) << 3;
  const float* p = base + (size_t)(row0 + r) * ld + kk + c0;
  v16h out;
#pragma unroll
  for (int i = 0; i < 8; ++i) {
    out[i]     = (_Float16)p[i];
    out[i + 8] = (_Float16)p[i + 16];
  }
  return out;
}

// 16-byte global -> LDS copy: async DMA on CDNA5, VGPR fallback otherwise.
static __device__ __forceinline__ void copy16_g2l(const _Float16* g, _Float16* l) {
#if HAVE_ASYNC
  __builtin_amdgcn_global_load_async_to_lds_b128(
      (AS1 v4i*)(uintptr_t)g, (AS3 v4i*)(uintptr_t)l, 0, 0);
#else
  *(v8h*)l = *(const v8h*)g;
#endif
}

static __device__ __forceinline__ void stage_join() {
#if HAVE_ASYNC
  __builtin_amdgcn_s_wait_asynccnt(0);
#endif
  __syncthreads();
}

// ---------------------------------------------------------------------------
__global__ void cast_f32_to_f16(const float* __restrict__ src,
                                _Float16* __restrict__ dst, int n) {
  int i = blockIdx.x * blockDim.x + threadIdx.x;
  const int stride = gridDim.x * blockDim.x;
  for (; i < n; i += stride) dst[i] = (_Float16)src[i];
}

// ---------------------------------------------------------------------------
// Y = A (MxK f16) * W^T (W: NxK f16).  128x128 block tile, 8 waves of 64x32
// (8 WMMAs / 32-k step), A/W slabs async-DMA'd to LDS, rows padded to 40
// halves for conflict-free ds_load_b128.
// STORE: 0 = f16 row-major MxN, 1 = f16 transposed NxM, 2 = f32 row-major.
// ---------------------------------------------------------------------------
template <int STORE>
__global__ void __launch_bounds__(256)
wmma_gemm_nt(const _Float16* __restrict__ A, const _Float16* __restrict__ W,
             void* __restrict__ Yv, int M, int N, int K) {
  __shared__ _Float16 As[128 * 40];
  __shared__ _Float16 Bs[128 * 40];
  const int lane = threadIdx.x & 31;
  const int wave = uni(threadIdx.x >> 5);
  const int nb   = N >> 7;
  const int bm   = blockIdx.x / nb;
  const int bn   = blockIdx.x - bm * nb;
  const int wm   = (wave >> 2) << 6;
  const int wn   = (wave & 3) << 5;
  const int gM0  = bm << 7, gN0 = bn << 7;

  v8f acc[4][2] = {};

  for (int kk = 0; kk < K; kk += 32) {
    for (int c = threadIdx.x; c < 512; c += 256) {
      const int row = c >> 2, col = (c & 3) << 3;
      copy16_g2l(A + (size_t)(gM0 + row) * K + kk + col, &As[row * 40 + col]);
      copy16_g2l(W + (size_t)(gN0 + row) * K + kk + col, &Bs[row * 40 + col]);
    }
    stage_join();

    v16h af[4], bf[2];
#pragma unroll
    for (int i = 0; i < 4; ++i) af[i] = load_frag_f16(As, 40, wm + (i << 4), 0, lane);
#pragma unroll
    for (int j = 0; j < 2; ++j) bf[j] = load_frag_f16(Bs, 40, wn + (j << 4), 0, lane);
#pragma unroll
    for (int i = 0; i < 4; ++i)
#pragma unroll
      for (int j = 0; j < 2; ++j)
        acc[i][j] = __builtin_amdgcn_wmma_f32_16x16x32_f16(
            false, af[i], false, bf[j], (short)0, acc[i][j], false, false);
    __syncthreads();
  }

  const int cn = lane & 15, rb = (lane >> 4) << 3;
#pragma unroll
  for (int i = 0; i < 4; ++i)
#pragma unroll
    for (int j = 0; j < 2; ++j) {
      const int n = gN0 + wn + (j << 4) + cn;
#pragma unroll
      for (int e = 0; e < 8; ++e) {
        const int m = gM0 + wm + (i << 4) + rb + e;
        if (STORE == 0)      ((_Float16*)Yv)[(size_t)m * N + n] = (_Float16)acc[i][j][e];
        else if (STORE == 1) ((_Float16*)Yv)[(size_t)n * M + m] = (_Float16)acc[i][j][e];
        else                 ((float*)Yv)[(size_t)m * N + n]    = acc[i][j][e];
      }
    }
}

// ---------------------------------------------------------------------------
// Fused scores + causal mask + softmax.  One block per (h, qt) 16-row strip.
// Wave w owns k-tiles [16w, 16w+16) => 256 columns, 16 v8f accumulators.
// Fully-masked tiles skip WMMA via a *scalar* branch (indices readfirstlaned).
// Row max/sum: shfl_xor butterfly across each 16-lane half (rows 0-7 in lanes
// 0-15, rows 8-15 in lanes 16-31), then 8x16 LDS cross-wave reduction.
// Writes normalized attn exactly once; masked region written as 0.
// ---------------------------------------------------------------------------
__global__ void __launch_bounds__(256)
attn_scores_softmax(const _Float16* __restrict__ Q, const _Float16* __restrict__ Km,
                    float* __restrict__ attn) {
  __shared__ float red[8][16];
  const int lane = threadIdx.x & 31;
  const int wave = uni(threadIdx.x >> 5);
  const int qt = uni(blockIdx.x & 127);
  const int h  = uni(blockIdx.x >> 7);
  const int q0 = qt << 4;
  const int cn = lane & 15, rb = (lane >> 4) << 3;

  const v16h a0 = load_frag_f16(Q, DDm, q0, h * DHH,      lane);
  const v16h a1 = load_frag_f16(Q, DDm, q0, h * DHH + 32, lane);

  v8f acc[16];
  float pmax[8];
#pragma unroll
  for (int e = 0; e < 8; ++e) pmax[e] = -3.4e38f;

#pragma unroll
  for (int t = 0; t < 16; ++t) {
    const int kt = (wave << 4) + t;
    if (kt <= qt) {                       // scalar branch: EXEC untouched
      v16h b0 = load_frag_f16(Km, DDm, kt << 4, h * DHH,      lane);
      v16h b1 = load_frag_f16(Km, DDm, kt << 4, h * DHH + 32, lane);
      v8f c = {};
      c = __builtin_amdgcn_wmma_f32_16x16x32_f16(false, a0, false, b0, (short)0, c, false, false);
      c = __builtin_amdgcn_wmma_f32_16x16x32_f16(false, a1, false, b1, (short)0, c, false, false);
#pragma unroll
      for (int e = 0; e < 8; ++e) {
        float s = c[e] * 0.125f;
        if ((kt << 4) + cn > q0 + rb + e) s = -1.0e30f;   // diagonal tile mask
        acc[t][e] = s;
        pmax[e] = fmaxf(pmax[e], s);
      }
    }
  }

  // row max: butterfly over the 16-lane half-group, then cross-wave via LDS
#pragma unroll
  for (int m = 1; m < 16; m <<= 1)
#pragma unroll
    for (int e = 0; e < 8; ++e) pmax[e] = fmaxf(pmax[e], __shfl_xor(pmax[e], m, 32));
  if ((lane & 15) == 0)
#pragma unroll
    for (int e = 0; e < 8; ++e) red[wave][rb + e] = pmax[e];
  __syncthreads();
  float rmax[8];
#pragma unroll
  for (int e = 0; e < 8; ++e) {
    float m = -3.4e38f;
#pragma unroll
    for (int w = 0; w < 8; ++w) m = fmaxf(m, red[w][rb + e]);
    rmax[e] = m;
  }
  __syncthreads();

  // exp + row sum
  float psum[8] = {0.f, 0.f, 0.f, 0.f, 0.f, 0.f, 0.f, 0.f};
#pragma unroll
  for (int t = 0; t < 16; ++t) {
    const int kt = (wave << 4) + t;
    if (kt <= qt) {
#pragma unroll
      for (int e = 0; e < 8; ++e) {
        const float ev = __expf(acc[t][e] - rmax[e]);
        acc[t][e] = ev;
        psum[e] += ev;
      }
    }
  }
#pragma unroll
  for (int m = 1; m < 16; m <<= 1)
#pragma unroll
    for (int e = 0; e < 8; ++e) psum[e] += __shfl_xor(psum[e], m, 32);
  if ((lane & 15) == 0)
#pragma unroll
    for (int e = 0; e < 8; ++e) red[wave][rb + e] = psum[e];
  __syncthreads();
  float rinv[8];
#pragma unroll
  for (int e = 0; e < 8; ++e) {
    float s = 0.f;
#pragma unroll
    for (int w = 0; w < 8; ++w) s += red[w][rb + e];
    rinv[e] = 1.0f / s;
  }

  // normalized write-out (masked tiles -> exact zeros)
  float* dst = attn + ((size_t)h * TT + q0) * TT;
#pragma unroll
  for (int t = 0; t < 16; ++t) {
    const int kt = (wave << 4) + t;
    if (kt <= qt) {
#pragma unroll
      for (int e = 0; e < 8; ++e)
        dst[(size_t)(rb + e) * TT + (kt << 4) + cn] = acc[t][e] * rinv[e];
    } else {
#pragma unroll
      for (int e = 0; e < 8; ++e)
        dst[(size_t)(rb + e) * TT + (kt << 4) + cn] = 0.f;
    }
  }
}

// ---------------------------------------------------------------------------
// OH[q, h*64+n] = sum_k attn[h,q,k] * V[k, h*64+n].  One wave per (h,qt),
// all 4 n-tiles resident; causal k-loop bound (skipped region is exact 0).
// ---------------------------------------------------------------------------
__global__ void __launch_bounds__(256)
attn_v(const float* __restrict__ attn, const _Float16* __restrict__ VT,
       _Float16* __restrict__ OH) {
  const int lane = threadIdx.x & 31;
  const int job = uni(blockIdx.x * 8 + (threadIdx.x >> 5));  // 16*128 jobs
  const int qt = job & 127;
  const int h  = job >> 7;
  if (h >= HH) return;

  const float* arows = attn + (size_t)h * TT * TT;
  v8f acc[4] = {};
  const int kend = (((qt << 4) + 16) + 31) & ~31;
  for (int kk = 0; kk < kend; kk += 32) {
    const v16h a = load_frag_f32cvt(arows, TT, qt << 4, kk, lane);
#pragma unroll
    for (int nt = 0; nt < 4; ++nt) {
      v16h b = load_frag_f16(VT, TT, h * DHH + (nt << 4), kk, lane);
      acc[nt] = __builtin_amdgcn_wmma_f32_16x16x32_f16(
          false, a, false, b, (short)0, acc[nt], false, false);
    }
  }
  const int cn = lane & 15, rb = (lane >> 4) << 3;
#pragma unroll
  for (int nt = 0; nt < 4; ++nt) {
    const int n = h * DHH + (nt << 4) + cn;
#pragma unroll
    for (int e = 0; e < 8; ++e)
      OH[(size_t)((qt << 4) + rb + e) * DDm + n] = (_Float16)acc[nt][e];
  }
}

// ---------------------------------------------------------------------------
extern "C" void kernel_launch(void* const* d_in, const int* in_sizes, int n_in,
                              void* d_out, int out_size, void* d_ws, size_t ws_size,
                              hipStream_t stream) {
  (void)in_sizes; (void)n_in; (void)out_size; (void)ws_size;
  const float* x  = (const float*)d_in[0];
  const float* Wq = (const float*)d_in[1];
  const float* Wk = (const float*)d_in[2];
  const float* Wv = (const float*)d_in[3];
  const float* Wo = (const float*)d_in[4];
  // d_in[5] = causal mask, applied analytically.

  const size_t TD = (size_t)TT * DDm;
  const size_t WW = (size_t)DDm * DDm;

  _Float16* x16  = (_Float16*)d_ws;
  _Float16* wq16 = x16  + TD;
  _Float16* wk16 = wq16 + WW;
  _Float16* wv16 = wk16 + WW;
  _Float16* wo16 = wv16 + WW;
  _Float16* Q16  = wo16 + WW;
  _Float16* K16  = Q16  + TD;
  _Float16* VT16 = K16  + TD;   // D x T (transposed V)
  _Float16* OH16 = VT16 + TD;

  float* out  = (float*)d_out;
  float* attn = out + TD;

  cast_f32_to_f16<<<512, 256, 0, stream>>>(x,  x16,  (int)TD);
  cast_f32_to_f16<<<512, 256, 0, stream>>>(Wq, wq16, (int)WW);
  cast_f32_to_f16<<<512, 256, 0, stream>>>(Wk, wk16, (int)WW);
  cast_f32_to_f16<<<512, 256, 0, stream>>>(Wv, wv16, (int)WW);
  cast_f32_to_f16<<<512, 256, 0, stream>>>(Wo, wo16, (int)WW);

  wmma_gemm_nt<0><<<128, 256, 0, stream>>>(x16, wq16, Q16,  TT, DDm, DDm);
  wmma_gemm_nt<0><<<128, 256, 0, stream>>>(x16, wk16, K16,  TT, DDm, DDm);
  wmma_gemm_nt<1><<<128, 256, 0, stream>>>(x16, wv16, VT16, TT, DDm, DDm);

  // one block per (h, qt): 16*128 = 2048 blocks
  attn_scores_softmax<<<2048, 256, 0, stream>>>(Q16, K16, attn);

  // 16*128 wave-jobs / 8 = 256 blocks
  attn_v<<<256, 256, 0, stream>>>(attn, VT16, OH16);

  wmma_gemm_nt<2><<<128, 256, 0, stream>>>(OH16, wo16, out, TT, DDm, DDm);
}
